// Seq2Seq_13116830122663
// MI455X (gfx1250) — compile-verified, matrix operand order checked by reading, not run on hardware
//
#include <hip/hip_runtime.h>
#include <math.h>

// ---------------- constants (match reference) ----------------
#define BB   512
#define SS   64
#define TT   32
#define EE   512
#define HH   1024
#define VV   128      // VIN == VOUT == 128

// ---------------- types ----------------
typedef __attribute__((ext_vector_type(16))) __bf16 v16bf;
typedef __attribute__((ext_vector_type(8)))  float  v8f;
typedef unsigned short ushort_t;

union Frag16 {
    v16bf v;
    uint4 q[2];
};

__device__ __forceinline__ ushort_t f32_to_bf16(float f) {
    unsigned int u = __float_as_uint(f);
    unsigned int r = 0x7FFFu + ((u >> 16) & 1u);
    return (ushort_t)((u + r) >> 16);
}

__device__ __forceinline__ float sigmoidf_(float x) {
    return 1.0f / (1.0f + __expf(-x));
}

// ---------------- f32 -> bf16 weight conversion ----------------
__global__ void cvt_bf16_kernel(const float* __restrict__ in,
                                ushort_t* __restrict__ out, int n) {
    int i = blockIdx.x * blockDim.x + threadIdx.x;
    if (i < n) out[i] = f32_to_bf16(in[i]);
}

// ---------------- zero init ----------------
__global__ void zero_f32_kernel(float* __restrict__ p, int n) {
    int i = blockIdx.x * blockDim.x + threadIdx.x;
    if (i < n) p[i] = 0.0f;
}

// ---------------- token init: tok[b] = tgt[b*TT] ----------------
__global__ void tok_init_kernel(const int* __restrict__ tgt, int* __restrict__ tok) {
    int b = blockIdx.x * blockDim.x + threadIdx.x;
    if (b < BB) tok[b] = tgt[b * TT];
}

// ---------------- concat h = [h_fwd | h_bwd] ----------------
__global__ void concat_h_kernel(const float* __restrict__ hf,
                                const float* __restrict__ hb,
                                float* __restrict__ hd) {
    int i = blockIdx.x * blockDim.x + threadIdx.x;
    if (i >= BB * 2 * HH) return;
    int m = i >> 11;          // /2048
    int j = i & 2047;
    hd[i] = (j < HH) ? hf[m * HH + j] : hb[m * HH + (j - HH)];
}

// ---------------- GRU elementwise update (in-place h) ----------------
// gi, gh: (B, 3*Hs), h: (B, Hs). hshift = log2(Hs).
__global__ void gru_update_kernel(float* __restrict__ h,
                                  const float* __restrict__ gi,
                                  const float* __restrict__ gh,
                                  int hshift) {
    int i = blockIdx.x * blockDim.x + threadIdx.x;
    int Hs = 1 << hshift;
    int m = i >> hshift;
    int j = i & (Hs - 1);
    const float* gim = gi + (size_t)m * 3 * Hs;
    const float* ghm = gh + (size_t)m * 3 * Hs;
    float r = sigmoidf_(gim[j] + ghm[j]);
    float z = sigmoidf_(gim[Hs + j] + ghm[Hs + j]);
    float n = tanhf(gim[2 * Hs + j] + r * ghm[2 * Hs + j]);
    float hv = h[i];
    h[i] = (1.0f - z) * n + z * hv;
}

// ---------------- argmax over a row of ncols (one wave32 per row) ----------------
__global__ void argmax_kernel(const float* __restrict__ logits, int ldr, int ncols,
                              int* __restrict__ tok) {
    int b = blockIdx.x;
    int lane = threadIdx.x;           // 32 threads
    const float* row = logits + (size_t)b * ldr;
    float best = -INFINITY;
    int bi = 0x7FFFFFFF;
    for (int j = lane; j < ncols; j += 32) {
        float v = row[j];
        if (v > best || (v == best && j < bi)) { best = v; bi = j; }
    }
    #pragma unroll
    for (int off = 16; off > 0; off >>= 1) {
        float ov = __shfl_down(best, off, 32);
        int   oi = __shfl_down(bi,   off, 32);
        if (ov > best || (ov == best && oi < bi)) { best = ov; bi = oi; }
    }
    if (lane == 0) tok[b] = bi;
}

// ---------------- WMMA bf16 GEMM: C = A @ W^T + bias ----------------
// A: (M, K)  — either f32 matrix (lda = row stride) converted to bf16 on the
//              fly, or GATHER mode: row m = emb_bf16[ idx[m*idx_stride] ] (row length K)
// W: (N, K) bf16 row-major (so C[m][n] = sum_k A[m][k]*W[n][k])
// C: f32, row stride ldc. bias: f32 (N).
// Block: 256 threads (8 wave32), tile M=128 x N=64, K step 32.
// Grid: (N/64, M/128). All dims are exact multiples for every call site.
#define LDA_T 48   // LDS row stride in bf16 elems (96 B, 16B-aligned rows & sub-offsets)

template <bool GATHER>
__global__ void gemm_bf16_kernel(const float* __restrict__ Af32, int lda,
                                 const ushort_t* __restrict__ emb,
                                 const int* __restrict__ idx, int idx_stride,
                                 const ushort_t* __restrict__ W,
                                 const float* __restrict__ bias,
                                 float* __restrict__ C, int ldc, int K) {
    __shared__ ushort_t As[128 * LDA_T];
    __shared__ ushort_t Ws[64 * LDA_T];

    const int tid  = threadIdx.x;
    const int lane = tid & 31;
    const int wave = tid >> 5;            // 0..7 -> m sub-tile
    const int lm   = lane & 15;           // m (A) / n (B) within 16
    const int k0   = (lane < 16) ? 0 : 8; // K-half select per ISA fragment layout

    const int m0 = blockIdx.y * 128;
    const int n0 = blockIdx.x * 64;

    // cooperative-load coordinates
    const int arow  = tid >> 1;           // 0..127
    const int ahalf = (tid & 1) * 16;     // 0 or 16 (bf16 elems)
    const int wrow  = tid >> 2;           // 0..63
    const int wseg  = (tid & 3) * 8;      // 0,8,16,24 (bf16 elems)

    v8f acc[4] = {};

    for (int kc = 0; kc < K; kc += 32) {
        // ---- stage A tile (128 x 32 bf16) ----
        if (GATHER) {
            int g = idx[(m0 + arow) * idx_stride];
            const ushort_t* src = emb + (size_t)g * K + kc + ahalf;
            *(uint4*)&As[arow * LDA_T + ahalf]     = *(const uint4*)(src);
            *(uint4*)&As[arow * LDA_T + ahalf + 8] = *(const uint4*)(src + 8);
        } else {
            const float* src = Af32 + (size_t)(m0 + arow) * lda + kc + ahalf;
            float4 f0 = *(const float4*)(src + 0);
            float4 f1 = *(const float4*)(src + 4);
            float4 f2 = *(const float4*)(src + 8);
            float4 f3 = *(const float4*)(src + 12);
            ushort_t t[16];
            t[0]=f32_to_bf16(f0.x); t[1]=f32_to_bf16(f0.y); t[2]=f32_to_bf16(f0.z); t[3]=f32_to_bf16(f0.w);
            t[4]=f32_to_bf16(f1.x); t[5]=f32_to_bf16(f1.y); t[6]=f32_to_bf16(f1.z); t[7]=f32_to_bf16(f1.w);
            t[8]=f32_to_bf16(f2.x); t[9]=f32_to_bf16(f2.y); t[10]=f32_to_bf16(f2.z); t[11]=f32_to_bf16(f2.w);
            t[12]=f32_to_bf16(f3.x); t[13]=f32_to_bf16(f3.y); t[14]=f32_to_bf16(f3.z); t[15]=f32_to_bf16(f3.w);
            *(uint4*)&As[arow * LDA_T + ahalf]     = *(uint4*)&t[0];
            *(uint4*)&As[arow * LDA_T + ahalf + 8] = *(uint4*)&t[8];
        }
        // ---- stage W tile (64 x 32 bf16) ----
        *(uint4*)&Ws[wrow * LDA_T + wseg] =
            *(const uint4*)(W + (size_t)(n0 + wrow) * K + kc + wseg);

        __syncthreads();

        // ---- WMMA: each wave does 16(M) x 64(N) ----
        Frag16 a;
        const ushort_t* ap = &As[(wave * 16 + lm) * LDA_T + k0];
        a.q[0] = *(const uint4*)(ap);
        a.q[1] = *(const uint4*)(ap + 16);
        #pragma unroll
        for (int s = 0; s < 4; ++s) {
            Frag16 bfrag;
            const ushort_t* bp = &Ws[(s * 16 + lm) * LDA_T + k0];
            bfrag.q[0] = *(const uint4*)(bp);
            bfrag.q[1] = *(const uint4*)(bp + 16);
            acc[s] = __builtin_amdgcn_wmma_f32_16x16x32_bf16(
                false, a.v, false, bfrag.v, (short)0, acc[s], false, false);
        }
        __syncthreads();
    }

    // ---- store: C/D fragment layout: VGPR r -> M = (lane<16?0:8)+r, N = lane%16 ----
    const int cn = lm;
    const int cm = (lane >> 4) * 8;
    #pragma unroll
    for (int s = 0; s < 4; ++s) {
        float bv = bias ? bias[n0 + s * 16 + cn] : 0.0f;
        #pragma unroll
        for (int r = 0; r < 8; ++r) {
            C[(size_t)(m0 + wave * 16 + cm + r) * ldc + (n0 + s * 16 + cn)] =
                acc[s][r] + bv;
        }
    }
}

// ============================================================
//                         host side
// ============================================================
extern "C" void kernel_launch(void* const* d_in, const int* in_sizes, int n_in,
                              void* d_out, int out_size, void* d_ws, size_t ws_size,
                              hipStream_t stream) {
    (void)in_sizes; (void)n_in; (void)out_size; (void)ws_size;

    // inputs in setup_inputs() order
    const int*   src     = (const int*)  d_in[0];   // (B, S)
    const int*   tgt     = (const int*)  d_in[1];   // (B, T)
    const float* enc_emb = (const float*)d_in[2];   // (VIN, E)
    const float* dec_emb = (const float*)d_in[3];   // (VOUT, E)
    const float* W_ih_f  = (const float*)d_in[4];   // (3H, E)
    const float* W_hh_f  = (const float*)d_in[5];   // (3H, H)
    const float* b_ih_f  = (const float*)d_in[6];
    const float* b_hh_f  = (const float*)d_in[7];
    const float* W_ih_b  = (const float*)d_in[8];
    const float* W_hh_b  = (const float*)d_in[9];
    const float* b_ih_b  = (const float*)d_in[10];
    const float* b_hh_b  = (const float*)d_in[11];
    const float* W_ih_d  = (const float*)d_in[12];  // (6H, E)
    const float* W_hh_d  = (const float*)d_in[13];  // (6H, 2H)
    const float* b_ih_d  = (const float*)d_in[14];
    const float* b_hh_d  = (const float*)d_in[15];
    const float* W_fc    = (const float*)d_in[16];  // (V, 2H)
    const float* b_fc    = (const float*)d_in[17];

    float* out = (float*)d_out;                     // (B, T-1, V)

    // ---- workspace bump allocator (deterministic layout) ----
    char*  ws  = (char*)d_ws;
    size_t off = 0;
    auto alloc = [&](size_t bytes) -> void* {
        void* p = ws + off;
        off = (off + bytes + 255) & ~(size_t)255;
        return p;
    };

    const int nWihE  = 3 * HH * EE;       // 1572864
    const int nWhhH  = 3 * HH * HH;       // 3145728
    const int nWihDe = 6 * HH * EE;       // 3145728
    const int nWhhDe = 6 * HH * 2 * HH;   // 12582912
    const int nWfc   = VV * 2 * HH;       // 262144
    const int nEmb   = VV * EE;           // 65536

    ushort_t* bWihF = (ushort_t*)alloc((size_t)nWihE  * 2);
    ushort_t* bWhhF = (ushort_t*)alloc((size_t)nWhhH  * 2);
    ushort_t* bWihB = (ushort_t*)alloc((size_t)nWihE  * 2);
    ushort_t* bWhhB = (ushort_t*)alloc((size_t)nWhhH  * 2);
    ushort_t* bWihD = (ushort_t*)alloc((size_t)nWihDe * 2);
    ushort_t* bWhhD = (ushort_t*)alloc((size_t)nWhhDe * 2);
    ushort_t* bWfc  = (ushort_t*)alloc((size_t)nWfc   * 2);
    ushort_t* bEncE = (ushort_t*)alloc((size_t)nEmb   * 2);
    ushort_t* bDecE = (ushort_t*)alloc((size_t)nEmb   * 2);

    float* h_fwd = (float*)alloc((size_t)BB * HH * 4);
    float* h_bwd = (float*)alloc((size_t)BB * HH * 4);
    float* h_dec = (float*)alloc((size_t)BB * 2 * HH * 4);
    float* gi    = (float*)alloc((size_t)BB * 6 * HH * 4);
    float* gh    = (float*)alloc((size_t)BB * 6 * HH * 4);
    int*   tok   = (int*)  alloc((size_t)BB * 4);

    const dim3 blk(256);
    auto cvt = [&](const float* in, ushort_t* o, int n) {
        cvt_bf16_kernel<<<dim3((n + 255) / 256), blk, 0, stream>>>(in, o, n);
    };

    // ---- weight / embedding conversion to bf16 ----
    cvt(W_ih_f, bWihF, nWihE);
    cvt(W_hh_f, bWhhF, nWhhH);
    cvt(W_ih_b, bWihB, nWihE);
    cvt(W_hh_b, bWhhB, nWhhH);
    cvt(W_ih_d, bWihD, nWihDe);
    cvt(W_hh_d, bWhhD, nWhhDe);
    cvt(W_fc,   bWfc,  nWfc);
    cvt(enc_emb, bEncE, nEmb);
    cvt(dec_emb, bDecE, nEmb);

    // ---- init hidden states ----
    zero_f32_kernel<<<dim3((BB * HH) / 256), blk, 0, stream>>>(h_fwd, BB * HH);
    zero_f32_kernel<<<dim3((BB * HH) / 256), blk, 0, stream>>>(h_bwd, BB * HH);

    const dim3 gridEnc(3 * HH / 64, BB / 128);   // (48, 4)
    const dim3 gridDec(6 * HH / 64, BB / 128);   // (96, 4)
    const dim3 gridFc (VV / 64,     BB / 128);   // (2, 4)

    // ---- encoder: 64 steps, forward + backward ----
    for (int s = 0; s < SS; ++s) {
        // forward direction, input column s
        gemm_bf16_kernel<true><<<gridEnc, blk, 0, stream>>>(
            nullptr, 0, bEncE, src + s, SS, bWihF, b_ih_f, gi, 3 * HH, EE);
        gemm_bf16_kernel<false><<<gridEnc, blk, 0, stream>>>(
            h_fwd, HH, nullptr, nullptr, 0, bWhhF, b_hh_f, gh, 3 * HH, HH);
        gru_update_kernel<<<dim3((BB * HH) / 256), blk, 0, stream>>>(h_fwd, gi, gh, 10);

        // backward direction, input column S-1-s
        gemm_bf16_kernel<true><<<gridEnc, blk, 0, stream>>>(
            nullptr, 0, bEncE, src + (SS - 1 - s), SS, bWihB, b_ih_b, gi, 3 * HH, EE);
        gemm_bf16_kernel<false><<<gridEnc, blk, 0, stream>>>(
            h_bwd, HH, nullptr, nullptr, 0, bWhhB, b_hh_b, gh, 3 * HH, HH);
        gru_update_kernel<<<dim3((BB * HH) / 256), blk, 0, stream>>>(h_bwd, gi, gh, 10);
    }

    // ---- h_dec = concat(h_fwd, h_bwd) ----
    concat_h_kernel<<<dim3((BB * 2 * HH) / 256), blk, 0, stream>>>(h_fwd, h_bwd, h_dec);
    tok_init_kernel<<<dim3(2), blk, 0, stream>>>(tgt, tok);

    // ---- decoder: 31 steps with greedy feedback ----
    for (int t = 0; t < TT - 1; ++t) {
        gemm_bf16_kernel<true><<<gridDec, blk, 0, stream>>>(
            nullptr, 0, bDecE, tok, 1, bWihD, b_ih_d, gi, 6 * HH, EE);
        gemm_bf16_kernel<false><<<gridDec, blk, 0, stream>>>(
            h_dec, 2 * HH, nullptr, nullptr, 0, bWhhD, b_hh_d, gh, 6 * HH, 2 * HH);
        gru_update_kernel<<<dim3((BB * 2 * HH) / 256), blk, 0, stream>>>(h_dec, gi, gh, 11);

        // logits straight into d_out at [b][t][*]  (row stride (T-1)*V)
        float* logits_t = out + (size_t)t * VV;
        gemm_bf16_kernel<false><<<gridFc, blk, 0, stream>>>(
            h_dec, 2 * HH, nullptr, nullptr, 0, bWfc, b_fc,
            logits_t, (TT - 1) * VV, 2 * HH);

        argmax_kernel<<<dim3(BB), dim3(32), 0, stream>>>(
            logits_t, (TT - 1) * VV, VV, tok);
    }
}
